// SlidingWindowAttention_62380105007291
// MI455X (gfx1250) — matrix operand order repaired and checked by reading, not compiled, not run on hardware
//
#include <hip/hip_runtime.h>

// ---------------------------------------------------------------------------
// SlidingWindowAttention for MI455X (gfx1250): bf16 WMMA + TDM + async-LDS.
// B=1, S=4096, D=1024, H=16, W=512, HD=64, C=8
// ---------------------------------------------------------------------------

typedef __attribute__((ext_vector_type(16))) __bf16 v16bf;
typedef __attribute__((ext_vector_type(8)))  float  v8f;
typedef __attribute__((ext_vector_type(4))) unsigned tdm_g0_t;
typedef __attribute__((ext_vector_type(8))) int      tdm_g1_t;
typedef __attribute__((ext_vector_type(4))) int      tdm_g2_t;
typedef __attribute__((ext_vector_type(8))) int      tdm_g4_t;

union AccU  { v8f v;  float    f[8]; };
union FragU { v16bf v; unsigned u[8]; };

__device__ __forceinline__ unsigned short f32_bf16(float f) {
  unsigned x = __float_as_uint(f);
  unsigned r = (x + 0x7FFFu + ((x >> 16) & 1u)) >> 16;   // round-to-nearest-even
  return (unsigned short)r;
}

__device__ __forceinline__ unsigned lds_off(const void* p) {
  // generic LDS address: addr[31:0] is the wave-relative LDS byte offset
  return (unsigned)(unsigned long long)p;
}

// --- CDNA5 async global->LDS copy (ASYNCcnt) -------------------------------
__device__ __forceinline__ void async_ld_b128(unsigned lds, const void* g) {
  asm volatile("global_load_async_to_lds_b128 %0, %1, off"
               :: "v"(lds), "v"(g) : "memory");
}
__device__ __forceinline__ void wait_async0() {
  asm volatile("s_wait_asynccnt 0x0" ::: "memory");
}

// --- CDNA5 Tensor Data Mover: 2D tile load with LDS padding (TENSORcnt) ----
// tile_d0 elements per row (contiguous), tile_d1 rows, row stride in elements.
// pad_icode: pad after 2^(c+1) dwords; pad_acode: pad (c+1) dwords.
__device__ __forceinline__ void tdm_load_2d(unsigned lds, const void* gaddr,
                                            int tile_d0, int tile_d1,
                                            int row_stride_elems,
                                            int pad_icode, int pad_acode) {
  unsigned long long ga = (unsigned long long)gaddr;
  tdm_g0_t g0;
  g0[0] = 1u;                                            // count=1 (valid D#)
  g0[1] = lds;                                           // lds_addr
  g0[2] = (unsigned)(ga & 0xFFFFFFFFu);                  // global_addr[31:0]
  g0[3] = (unsigned)((ga >> 32) & 0x01FFFFFFu) | (2u << 30);  // addr hi | type=2
  tdm_g1_t g1;
  unsigned w0 = (1u << 16)                               // data_size = 2 bytes
              | (1u << 20)                               // pad_enable
              | ((unsigned)pad_icode << 22)
              | ((unsigned)pad_acode << 25);
  g1[0] = (int)w0;
  g1[1] = (int)(((unsigned)tile_d0 & 0xFFFFu) << 16);    // tensor_dim0 lo16
  g1[2] = (int)(((unsigned)tile_d1 & 0xFFFFu) << 16);    // dim0 hi=0 | tensor_dim1 lo16
  g1[3] = (int)(((unsigned)tile_d0 & 0xFFFFu) << 16);    // dim1 hi=0 | tile_dim0
  g1[4] = (int)((unsigned)tile_d1 & 0xFFFFu);            // tile_dim1 (tile_dim2=0)
  g1[5] = row_stride_elems;                              // tensor_dim0_stride lo32
  g1[6] = 0;                                             // stride hi | dim1_stride
  g1[7] = 0;
  tdm_g2_t gz = {0, 0, 0, 0};
  tdm_g4_t gz8 = {0, 0, 0, 0, 0, 0, 0, 0};
  __builtin_amdgcn_tensor_load_to_lds(g0, g1, gz, gz, gz8, 0);
}
__device__ __forceinline__ void wait_tensor0() {
  __builtin_amdgcn_s_wait_tensorcnt(0);
}

// --- WMMA fragment loaders (contiguous k = 16h + s; A and B match) ----------------------------
// A 16x32 bf16: lane m=l&15, half=l>>4; VGPR v holds pair K=16h+2v
__device__ __forceinline__ v16bf load_frag_a(const unsigned short* s, int stride) {
  const int lane = threadIdx.x & 31;
  const int m = lane & 15, h = lane >> 4;
  const unsigned short* row = s + m * stride;
  FragU f;
#pragma unroll
  for (int v = 0; v < 8; ++v) {
    int kb = h * 16 + v * 2;
    f.u[v] = *(const unsigned*)(row + kb);
  }
  return f.v;
}
// B 32x16 bf16 from an n-major (K-contiguous) tile: col n=l&15, K=16h+2v
__device__ __forceinline__ v16bf load_frag_b(const unsigned short* s, int stride) {
  const int lane = threadIdx.x & 31;
  const int n = lane & 15, h = lane >> 4;
  const unsigned short* row = s + n * stride + h * 16;
  FragU f;
#pragma unroll
  for (int v = 0; v < 8; ++v) f.u[v] = *(const unsigned*)(row + v * 2);
  return f.v;
}
// B 32x16 bf16 from a K-major (row-major B) tile: gather 16-bit pairs
__device__ __forceinline__ v16bf load_frag_b_kmajor(const unsigned short* s, int stride) {
  const int lane = threadIdx.x & 31;
  const int n = lane & 15, h = lane >> 4;
  FragU f;
#pragma unroll
  for (int v = 0; v < 8; ++v) {
    unsigned lo = s[(h * 16 + 2 * v + 0) * stride + n];
    unsigned hi = s[(h * 16 + 2 * v + 1) * stride + n];
    f.u[v] = lo | (hi << 16);
  }
  return f.v;
}

// ---------------------------------------------------------------------------
// fp32 -> bf16 conversion
// ---------------------------------------------------------------------------
__global__ void swa_cvt_bf16_kernel(const float* __restrict__ in,
                                    unsigned short* __restrict__ out, int n) {
  int i = blockIdx.x * blockDim.x + threadIdx.x;
  if (i < n) out[i] = f32_bf16(in[i]);
}

// ---------------------------------------------------------------------------
// bf16 GEMM: C[M,N] = A[M,K]*B[K,N], fp32 accumulate. 128 thr (4 waves),
// tile 64x64, K-step 32. A tile staged via global_load_async_to_lds_b128.
// ---------------------------------------------------------------------------
#define GSTR 48  // 32 + 16 pad; 96-byte rows -> 16B-aligned for B128

__global__ void __launch_bounds__(128)
swa_gemm_bf16_kernel(const unsigned short* __restrict__ A,
                     const unsigned short* __restrict__ B,
                     unsigned short* __restrict__ outb,
                     float* __restrict__ outf,
                     int M, int N, int K, int store_f32) {
  __shared__ __align__(16) unsigned short As[64 * GSTR];
  __shared__ __align__(16) unsigned short Bt[64 * GSTR];

  const int row0 = blockIdx.y * 64;
  const int n0   = blockIdx.x * 64;
  const int tid  = threadIdx.x;
  const int w    = tid >> 5;

  AccU acc[4];
#pragma unroll
  for (int i = 0; i < 4; ++i)
#pragma unroll
    for (int r = 0; r < 8; ++r) acc[i].f[r] = 0.f;

  for (int k0 = 0; k0 < K; k0 += 32) {
    // A tile 64x32 bf16 (64B/row): async DMA to LDS, 16B chunks
#pragma unroll
    for (int it = 0; it < 2; ++it) {
      int idx = tid + it * 128;          // 256 chunks
      int m = idx >> 2, c4 = idx & 3;
      async_ld_b128(lds_off(As + m * GSTR + c4 * 8),
                    A + (size_t)(row0 + m) * K + k0 + c4 * 8);
    }
    // B tile 32x64 -> n-major transpose (batched regs then LDS stores)
    unsigned tb[8];
#pragma unroll
    for (int it = 0; it < 8; ++it) {
      int idx = tid + it * 128;
      int k = idx >> 5, n2 = idx & 31;
      tb[it] = *(const unsigned*)(B + (size_t)(k0 + k) * N + n0 + n2 * 2);
    }
#pragma unroll
    for (int it = 0; it < 8; ++it) {
      int idx = tid + it * 128;
      int k = idx >> 5, n2 = idx & 31;
      Bt[(n2 * 2 + 0) * GSTR + k] = (unsigned short)(tb[it] & 0xFFFFu);
      Bt[(n2 * 2 + 1) * GSTR + k] = (unsigned short)(tb[it] >> 16);
    }
    if (k0 + 32 < K) {
      __builtin_prefetch(B + (size_t)(k0 + 32 + (tid >> 2)) * N + n0, 0, 0);
    }
    wait_async0();
    __syncthreads();

    v16bf a = load_frag_a(As + (w * 16) * GSTR, GSTR);
#pragma unroll
    for (int nb = 0; nb < 4; ++nb) {
      v16bf b = load_frag_b(Bt + (nb * 16) * GSTR, GSTR);
      acc[nb].v = __builtin_amdgcn_wmma_f32_16x16x32_bf16(
          false, a, false, b, (short)0, acc[nb].v, false, false);
    }
    __syncthreads();
  }

  const int lane = tid & 31, ncol = lane & 15, half = lane >> 4;
#pragma unroll
  for (int nb = 0; nb < 4; ++nb)
#pragma unroll
    for (int r = 0; r < 8; ++r) {
      int row = row0 + w * 16 + r + 8 * half;
      int col = n0 + nb * 16 + ncol;
      if (store_f32) outf[(size_t)row * N + col] = acc[nb].f[r];
      else           outb[(size_t)row * N + col] = f32_bf16(acc[nb].f[r]);
    }
}

// ---------------------------------------------------------------------------
// Chunked sliding-window attention, flash-style online softmax.
// Grid: (qtile=8, chunk=8, head=16), 128 threads (4 waves).
// Q/K/V 64x64 bf16 tiles staged via TDM with LDS padding to stride 72.
// ---------------------------------------------------------------------------
#define W_SZ 512
#define QSTR 72  // 64 + 4-dword TDM pad; 144-byte rows (16B aligned)

__global__ void __launch_bounds__(128)
swa_attention_wmma_kernel(const unsigned short* __restrict__ qkv,
                          unsigned short* __restrict__ attn) {
  __shared__ __align__(16) unsigned short Qs[64 * QSTR];
  __shared__ __align__(16) unsigned short Ks[64 * QSTR];  // row=key, col=d
  __shared__ __align__(16) unsigned short Vs[64 * QSTR];  // row=key, col=d
  __shared__ __align__(16) unsigned short Ps[64 * QSTR];  // probs, row-major

  const int qt = blockIdx.x;           // query tile in chunk (0..7)
  const int c  = blockIdx.y;           // chunk (0..7)
  const int h  = blockIdx.z;           // head (0..15)
  const int tid = threadIdx.x;
  const int w   = tid >> 5;
  const int lane = tid & 31, ncol = lane & 15, half = lane >> 4;
  const int s0 = c * W_SZ + qt * 64;
  const float scale = 0.125f;          // 1/sqrt(64)

  // TDM: Q tile (64 rows x 64 elems, row stride 3072), pad 32dw->+4dw
  if (tid < 32) {
    tdm_load_2d(lds_off(Qs), qkv + (size_t)s0 * 3072 + h * 64,
                64, 64, 3072, /*icode 32dw*/4, /*acode 4dw*/3);
  }

  float Mrow[8], Lrow[8];
  AccU oacc[4];
#pragma unroll
  for (int r = 0; r < 8; ++r) { Mrow[r] = -1e30f; Lrow[r] = 0.f; }
#pragma unroll
  for (int i = 0; i < 4; ++i)
#pragma unroll
    for (int r = 0; r < 8; ++r) oacc[i].f[r] = 0.f;

  const int kb_lo = (c == 0) ? 8 : 0;  // chunk 0 has no previous chunk
  const int kb_hi = 8 + qt;            // causal upper bound (inclusive)

  for (int kb = kb_lo; kb <= kb_hi; ++kb) {
    const int kbase = (c - 1) * W_SZ + kb * 64;
    if (tid < 32) {
      tdm_load_2d(lds_off(Ks), qkv + (size_t)kbase * 3072 + 1024 + h * 64,
                  64, 64, 3072, 4, 3);
      tdm_load_2d(lds_off(Vs), qkv + (size_t)kbase * 3072 + 2048 + h * 64,
                  64, 64, 3072, 4, 3);
      wait_tensor0();                  // also covers the Q tile (same counter)
    }
    __syncthreads();

    // S = Q @ K^T  (16 rows per wave x 64 keys); Ks rows are keys -> B^T layout
    AccU sacc[4];
#pragma unroll
    for (int i = 0; i < 4; ++i)
#pragma unroll
      for (int r = 0; r < 8; ++r) sacc[i].f[r] = 0.f;
#pragma unroll
    for (int ks = 0; ks < 64; ks += 32) {
      v16bf a = load_frag_a(Qs + (w * 16) * QSTR + ks, QSTR);
#pragma unroll
      for (int nb = 0; nb < 4; ++nb) {
        v16bf b = load_frag_b(Ks + (nb * 16) * QSTR + ks, QSTR);
        sacc[nb].v = __builtin_amdgcn_wmma_f32_16x16x32_bf16(
            false, a, false, b, (short)0, sacc[nb].v, false, false);
      }
    }

    // scale + causal mask (kpos >= 0 guaranteed by kb_lo)
#pragma unroll
    for (int nb = 0; nb < 4; ++nb)
#pragma unroll
      for (int r = 0; r < 8; ++r) {
        int qpos = s0 + w * 16 + r + 8 * half;
        int kpos = kbase + nb * 16 + ncol;
        float sv = sacc[nb].f[r] * scale;
        sacc[nb].f[r] = (kpos <= qpos) ? sv : -1e30f;
      }

    // online softmax: rows live across the 16 lanes of each half-wave
#pragma unroll
    for (int r = 0; r < 8; ++r) {
      float mx = fmaxf(fmaxf(sacc[0].f[r], sacc[1].f[r]),
                       fmaxf(sacc[2].f[r], sacc[3].f[r]));
#pragma unroll
      for (int off = 1; off < 16; off <<= 1)
        mx = fmaxf(mx, __shfl_xor(mx, off, 32));
      float Mn = fmaxf(Mrow[r], mx);
      float alpha = __expf(Mrow[r] - Mn);
      float rs = 0.f;
#pragma unroll
      for (int nb = 0; nb < 4; ++nb) {
        float p = __expf(sacc[nb].f[r] - Mn);
        sacc[nb].f[r] = p;
        rs += p;
      }
#pragma unroll
      for (int off = 1; off < 16; off <<= 1) rs += __shfl_xor(rs, off, 32);
      Lrow[r] = Lrow[r] * alpha + rs;
      Mrow[r] = Mn;
#pragma unroll
      for (int db = 0; db < 4; ++db) oacc[db].f[r] *= alpha;
    }

    // restage P (C-layout) -> row-major bf16 in LDS (per-wave private region)
#pragma unroll
    for (int nb = 0; nb < 4; ++nb)
#pragma unroll
      for (int r = 0; r < 8; ++r)
        Ps[(w * 16 + r + 8 * half) * QSTR + nb * 16 + ncol] =
            f32_bf16(sacc[nb].f[r]);

    // O += P @ V  (V staged row-major [key][d] -> K-major B fragments)
#pragma unroll
    for (int ks = 0; ks < 64; ks += 32) {
      v16bf pa = load_frag_a(Ps + (w * 16) * QSTR + ks, QSTR);
#pragma unroll
      for (int db = 0; db < 4; ++db) {
        v16bf vb = load_frag_b_kmajor(Vs + ks * QSTR + db * 16, QSTR);
        oacc[db].v = __builtin_amdgcn_wmma_f32_16x16x32_bf16(
            false, pa, false, vb, (short)0, oacc[db].v, false, false);
      }
    }
    __syncthreads();
  }

  // epilogue: normalize, write heads-interleaved [s][h*64+d] as bf16
#pragma unroll
  for (int db = 0; db < 4; ++db)
#pragma unroll
    for (int r = 0; r < 8; ++r) {
      int srow = s0 + w * 16 + r + 8 * half;
      int d = db * 16 + ncol;
      attn[(size_t)srow * 1024 + h * 64 + d] =
          f32_bf16(oacc[db].f[r] / Lrow[r]);
    }
}

// ---------------------------------------------------------------------------
// launch
// ---------------------------------------------------------------------------
extern "C" void kernel_launch(void* const* d_in, const int* in_sizes, int n_in,
                              void* d_out, int out_size, void* d_ws, size_t ws_size,
                              hipStream_t stream) {
  (void)in_sizes; (void)n_in; (void)out_size; (void)ws_size;
  const float* x     = (const float*)d_in[0];  // [4096,1024]
  const float* w_qkv = (const float*)d_in[1];  // [1024,3072]
  const float* w_out = (const float*)d_in[2];  // [1024,1024]
  float* out = (float*)d_out;                  // [4096,1024]

  const size_t NX   = (size_t)4096 * 1024;
  const size_t NWQ  = (size_t)1024 * 3072;
  const size_t NWO  = (size_t)1024 * 1024;
  const size_t NQKV = (size_t)4096 * 3072;

  unsigned short* x_b    = (unsigned short*)d_ws;
  unsigned short* wqkv_b = x_b + NX;
  unsigned short* wout_b = wqkv_b + NWQ;
  unsigned short* qkv_b  = wout_b + NWO;
  unsigned short* attn_b = qkv_b + NQKV;       // total ~48 MiB of workspace

  swa_cvt_bf16_kernel<<<(int)((NX + 255) / 256), 256, 0, stream>>>(x, x_b, (int)NX);
  swa_cvt_bf16_kernel<<<(int)((NWQ + 255) / 256), 256, 0, stream>>>(w_qkv, wqkv_b, (int)NWQ);
  swa_cvt_bf16_kernel<<<(int)((NWO + 255) / 256), 256, 0, stream>>>(w_out, wout_b, (int)NWO);

  // qkv = x @ w_qkv   (M=4096, N=3072, K=1024) -> bf16
  swa_gemm_bf16_kernel<<<dim3(3072 / 64, 4096 / 64), 128, 0, stream>>>(
      x_b, wqkv_b, qkv_b, nullptr, 4096, 3072, 1024, 0);

  // attention per (qtile, chunk, head)
  swa_attention_wmma_kernel<<<dim3(8, 8, 16), 128, 0, stream>>>(qkv_b, attn_b);

  // out = attn @ w_out (M=4096, N=1024, K=1024) -> fp32
  swa_gemm_bf16_kernel<<<dim3(1024 / 64, 4096 / 64), 128, 0, stream>>>(
      attn_b, wout_b, nullptr, out, 4096, 1024, 1024, 1);
}
